// SurrogateNetwork_10385230922213
// MI455X (gfx1250) — matrix operand order repaired and checked by reading, not compiled, not run on hardware
//
#include <hip/hip_runtime.h>
#include <hip/hip_bf16.h>
#include <stdint.h>

typedef __attribute__((ext_vector_type(16))) _Float16 v16h;
typedef __attribute__((ext_vector_type(8)))  _Float16 v8h;
typedef __attribute__((ext_vector_type(8)))  float    v8f;

#define HDIM       128
#define HEADS      8
#define NEG_SLOPE  0.2f
#define LN_EPS     1e-5f

// ---------------- float <-> monotone-uint for atomic max over floats -------
__device__ __forceinline__ unsigned f2ord(float f) {
    unsigned u = __float_as_uint(f);
    return (u >> 31) ? ~u : (u | 0x80000000u);
}
__device__ __forceinline__ float ord2f(unsigned u) {
    return __uint_as_float((u >> 31) ? (u ^ 0x80000000u) : ~u);
}

// ---------------- input projection: h = x @ W_in + b_in (K=2, VALU) --------
__global__ void input_proj_kernel(const float* __restrict__ x,
                                  const float* __restrict__ Win,
                                  const float* __restrict__ bin,
                                  float* __restrict__ h,
                                  _Float16* __restrict__ hh, int N) {
    int t = blockIdx.x * blockDim.x + threadIdx.x;
    if (t >= N * HDIM) return;
    int n = t >> 7, j = t & 127;
    float v = x[2 * n] * Win[j] + x[2 * n + 1] * Win[HDIM + j] + bin[j];
    h[t]  = v;
    hh[t] = (_Float16)v;
}

// ---------------- f32 -> f16 convert (plain, for activations) --------------
__global__ void cvt16_kernel(const float* __restrict__ a,
                             _Float16* __restrict__ b, int n) {
    int t = blockIdx.x * blockDim.x + threadIdx.x;
    if (t < n) b[t] = (_Float16)a[t];
}

// ---------------- f32 weight -> f16 swizzled into WMMA B-fragment order ----
// Wsw[t*Ncols*16 + c*16 + j] = W[(t*16+j)*Ncols + c]; a lane's 16 B halves
// for (k-tile, column) are then 32 contiguous bytes -> 2x ds_load_b128.
__global__ void cvt16_swz_kernel(const float* __restrict__ W,
                                 _Float16* __restrict__ Wsw,
                                 int K, int Ncols) {
    int i = blockIdx.x * blockDim.x + threadIdx.x;
    if (i >= K * Ncols) return;
    int tile = i / (Ncols * 16);
    int rem  = i - tile * (Ncols * 16);
    int c = rem >> 4, j = rem & 15;
    Wsw[i] = (_Float16)W[(tile * 16 + j) * Ncols + c];
}

// ---------------- WMMA GEMM: C[M,NCOLS] = A[M,128] @ B[128,NCOLS] ----------
// A fp16 row-major, B fp16 pre-swizzled; C fp32 (+bias, +relu). Entire B is
// staged into LDS via global_load_async_to_lds_b128 (ASYNCcnt). One 16x16 C
// tile per wave, K=128 fully unrolled -> 4 v_wmma per tile.
template <int NCOLS>
__global__ __launch_bounds__(256)
void gemm_wmma_kernel(const _Float16* __restrict__ A,
                      const _Float16* __restrict__ Bsw,
                      const float* __restrict__ bias,
                      float* __restrict__ C,
                      int M, int doRelu) {
    constexpr int K = 128;
    constexpr int TOTAL = K * NCOLS;
    __shared__ __align__(16) _Float16 Blds[TOTAL];

    int tid = threadIdx.x;
    // async-copy whole (swizzled) B into LDS: 16B per lane per issue
#pragma unroll
    for (int i = tid * 8; i < TOTAL; i += 256 * 8) {
        unsigned ldsOff = (unsigned)(uintptr_t)(&Blds[i]);
        asm volatile("global_load_async_to_lds_b128 %0, %1, off"
                     :: "v"(ldsOff), "v"(Bsw + i) : "memory");
    }
    asm volatile("s_wait_asynccnt 0x0" ::: "memory");
    __syncthreads();

    int wave  = tid >> 5;
    int lane  = tid & 31;
    int row16 = lane & 15;
    int hi    = lane >> 4;                  // lane half selects K sub-range
    int rowBase = blockIdx.x << 4;
    int colBase = wave << 4;
    if (colBase >= NCOLS) return;           // wave-uniform: EXEC stays full

    int col  = colBase + row16;
    int arow = rowBase + row16;
    if (arow >= M) arow = M - 1;            // clamp keeps EXEC all-ones
    const _Float16* Arow = A + (size_t)arow * K;

    v8f acc = {};
#pragma unroll
    for (int k0 = 0; k0 < K; k0 += 32) {
        // A fragment 16x32 f16: half j -> K=hi*16+j (matches B order):
        // one contiguous 32B span as two 16B chunks.
        v8h a0 = *(const v8h*)(Arow + k0 + hi * 16);
        v8h a1 = *(const v8h*)(Arow + k0 + hi * 16 + 8);
        v16h a = __builtin_shufflevector(a0, a1, 0, 1, 2, 3, 4, 5, 6, 7,
                                                 8, 9, 10, 11, 12, 13, 14, 15);
        if (k0 + 32 < K)
            __builtin_prefetch(Arow + k0 + 64, 0, 3);   // global_prefetch_b8

        // B fragment 32x16 f16: half j -> K=k0+hi*16+j at column col;
        // swizzled layout makes this 32 contiguous LDS bytes.
        const v8h* Bp = (const v8h*)&Blds[(((k0 >> 4) + hi) * NCOLS + col) * 16];
        v16h b = __builtin_shufflevector(Bp[0], Bp[1], 0, 1, 2, 3, 4, 5, 6, 7,
                                                       8, 9, 10, 11, 12, 13, 14, 15);

        acc = __builtin_amdgcn_wmma_f32_16x16x32_f16(
                  false, a, false, b, (short)0, acc, false, false);
    }

    // C/D layout: VGPR r -> row M = hi*8 + r, lane -> col N = row16
    float bv = bias ? bias[col] : 0.0f;
#pragma unroll
    for (int r = 0; r < 8; ++r) {
        int orow = rowBase + hi * 8 + r;
        if (orow < M) {
            float v = acc[r] + bv;
            if (doRelu) v = fmaxf(v, 0.0f);
            C[(size_t)orow * NCOLS + col] = v;
        }
    }
}

// ---------------- per-(node,head) attention logits -------------------------
__global__ void alpha_kernel(const float* __restrict__ h2,
                             const float* __restrict__ a_src,
                             const float* __restrict__ a_dst,
                             float* __restrict__ as_out,
                             float* __restrict__ ad_out, int N) {
    int t = blockIdx.x * blockDim.x + threadIdx.x;
    if (t >= N * HEADS) return;
    int n = t >> 3, hd = t & 7;
    const float* hp = h2 + (size_t)n * HDIM + hd * 16;
    const float* sp = a_src + hd * 16;
    const float* dp = a_dst + hd * 16;
    float s1 = 0.0f, s2 = 0.0f;
#pragma unroll
    for (int d = 0; d < 16; ++d) { float v = hp[d]; s1 += v * sp[d]; s2 += v * dp[d]; }
    as_out[t] = s1;
    ad_out[t] = s2;
}

__global__ void init_maxv_kernel(unsigned* __restrict__ m, int n) {
    int t = blockIdx.x * blockDim.x + threadIdx.x;
    if (t < n) m[t] = 0x007FFFFFu;               // f2ord(-inf)
}

// ---------------- edge pass 1: segment max ---------------------------------
__global__ void edge_max_kernel(const int* __restrict__ ei, int E, int N,
                                const float* __restrict__ as,
                                const float* __restrict__ ad,
                                unsigned* __restrict__ maxv) {
    int t = blockIdx.x * blockDim.x + threadIdx.x;
    if (t >= (E + N) * HEADS) return;
    int e = t >> 3, hd = t & 7;
    int s, d;
    if (e < E) { s = ei[e]; d = ei[E + e]; } else { s = d = e - E; }
    float v = as[s * HEADS + hd] + ad[d * HEADS + hd];
    v = v > 0.0f ? v : v * NEG_SLOPE;
    atomicMax(maxv + d * HEADS + hd, f2ord(v));
}

// ---------------- edge pass 2: exp-sum denominator -------------------------
__global__ void edge_sum_kernel(const int* __restrict__ ei, int E, int N,
                                const float* __restrict__ as,
                                const float* __restrict__ ad,
                                const unsigned* __restrict__ maxv,
                                float* __restrict__ denom) {
    int t = blockIdx.x * blockDim.x + threadIdx.x;
    if (t >= (E + N) * HEADS) return;
    int e = t >> 3, hd = t & 7;
    int s, d;
    if (e < E) { s = ei[e]; d = ei[E + e]; } else { s = d = e - E; }
    float v = as[s * HEADS + hd] + ad[d * HEADS + hd];
    v = v > 0.0f ? v : v * NEG_SLOPE;
    float ex = __expf(v - ord2f(maxv[d * HEADS + hd]));
    atomicAdd(denom + d * HEADS + hd, ex);
}

// ---------------- edge pass 3: weighted scatter ----------------------------
__global__ void edge_scatter_kernel(const int* __restrict__ ei, int E, int N,
                                    const float* __restrict__ as,
                                    const float* __restrict__ ad,
                                    const unsigned* __restrict__ maxv,
                                    const float* __restrict__ denom,
                                    const float* __restrict__ h2,
                                    float* __restrict__ agg) {
    int t = blockIdx.x * blockDim.x + threadIdx.x;
    if (t >= (E + N) * HEADS) return;
    int e = t >> 3, hd = t & 7;
    int s, d;
    if (e < E) { s = ei[e]; d = ei[E + e]; } else { s = d = e - E; }
    float v = as[s * HEADS + hd] + ad[d * HEADS + hd];
    v = v > 0.0f ? v : v * NEG_SLOPE;
    float ex = __expf(v - ord2f(maxv[d * HEADS + hd]));
    float alpha = ex / denom[d * HEADS + hd];
    const float* srcp = h2 + (size_t)s * HDIM + hd * 16;
    float* dstp = agg + (size_t)d * HDIM + hd * 16;
    __builtin_prefetch(srcp, 0, 3);
#pragma unroll
    for (int k = 0; k < 16; ++k) atomicAdd(dstp + k, srcp[k] * alpha);
}

// ---------------- bias + relu + residual + layernorm (one wave per node) ---
__global__ __launch_bounds__(256)
void post_kernel(const float* __restrict__ agg,
                 const float* __restrict__ gbias,
                 const float* __restrict__ g,
                 const float* __restrict__ beta,
                 float* __restrict__ h,            // in: residual, out: new h
                 _Float16* __restrict__ hh, int N) {
    int wave = threadIdx.x >> 5;
    int lane = threadIdx.x & 31;
    int n = blockIdx.x * (blockDim.x >> 5) + wave;
    if (n >= N) return;                           // wave-uniform
    size_t base = (size_t)n * HDIM;
    float v[4];
    float s = 0.0f;
#pragma unroll
    for (int i = 0; i < 4; ++i) {
        int j = lane * 4 + i;
        float t = agg[base + j] + gbias[j];
        t = fmaxf(t, 0.0f);
        t += h[base + j];
        v[i] = t; s += t;
    }
#pragma unroll
    for (int m = 16; m > 0; m >>= 1) s += __shfl_xor(s, m, 32);
    float mu = s * (1.0f / 128.0f);
    float q = 0.0f;
#pragma unroll
    for (int i = 0; i < 4; ++i) { float dd = v[i] - mu; q += dd * dd; }
#pragma unroll
    for (int m = 16; m > 0; m >>= 1) q += __shfl_xor(q, m, 32);
    float inv = rsqrtf(q * (1.0f / 128.0f) + LN_EPS);
#pragma unroll
    for (int i = 0; i < 4; ++i) {
        int j = lane * 4 + i;
        float y = (v[i] - mu) * inv * g[j] + beta[j];
        h[base + j]  = y;
        hh[base + j] = (_Float16)y;
    }
}

// ---------------- final head: out = g2[N,64] @ W3[64] + b3 -----------------
__global__ void head_kernel(const float* __restrict__ g2,
                            const float* __restrict__ W3,
                            const float* __restrict__ b3,
                            float* __restrict__ out, int N) {
    int n = blockIdx.x * blockDim.x + threadIdx.x;
    if (n >= N) return;
    const float* gp = g2 + (size_t)n * 64;
    float s = b3[0];
#pragma unroll
    for (int k = 0; k < 64; ++k) s += gp[k] * W3[k];
    out[n] = s;
}

// ===========================================================================
extern "C" void kernel_launch(void* const* d_in, const int* in_sizes, int n_in,
                              void* d_out, int out_size, void* d_ws, size_t ws_size,
                              hipStream_t stream) {
    const float* x    = (const float*)d_in[0];
    const int*   ei   = (const int*)d_in[1];
    // d_in[2] node_types: unused by the reference
    const float* W_in = (const float*)d_in[3];
    const float* b_in = (const float*)d_in[4];
    const float* gatW[3]  = {(const float*)d_in[5],  (const float*)d_in[11], (const float*)d_in[17]};
    const float* gatAS[3] = {(const float*)d_in[6],  (const float*)d_in[12], (const float*)d_in[18]};
    const float* gatAD[3] = {(const float*)d_in[7],  (const float*)d_in[13], (const float*)d_in[19]};
    const float* gatB[3]  = {(const float*)d_in[8],  (const float*)d_in[14], (const float*)d_in[20]};
    const float* lnG[3]   = {(const float*)d_in[9],  (const float*)d_in[15], (const float*)d_in[21]};
    const float* lnB[3]   = {(const float*)d_in[10], (const float*)d_in[16], (const float*)d_in[22]};
    const float* W1 = (const float*)d_in[23]; const float* b1 = (const float*)d_in[24];
    const float* W2 = (const float*)d_in[25]; const float* b2 = (const float*)d_in[26];
    const float* W3 = (const float*)d_in[27]; const float* b3 = (const float*)d_in[28];

    const int N = in_sizes[0] / 2;
    const int E = in_sizes[1] / 2;

    // ---- carve workspace ----
    char* p = (char*)d_ws;
    auto carve = [&](size_t bytes) -> void* {
        void* r = (void*)p; p += (bytes + 255) & ~(size_t)255; return r;
    };
    float*    h_f32 = (float*)   carve((size_t)N * HDIM * 4);
    _Float16* h_f16 = (_Float16*)carve((size_t)N * HDIM * 2);
    float*    h2    = (float*)   carve((size_t)N * HDIM * 4);
    float*    agg   = (float*)   carve((size_t)N * HDIM * 4);
    float*    asb   = (float*)   carve((size_t)N * HEADS * 4);
    float*    adb   = (float*)   carve((size_t)N * HEADS * 4);
    unsigned* maxv  = (unsigned*)carve((size_t)N * HEADS * 4);
    float*    denom = (float*)   carve((size_t)N * HEADS * 4);
    _Float16* Wh    = (_Float16*)carve(128 * 128 * 2);
    _Float16* g1h   = (_Float16*)agg;     // reuse (free after last GAT layer)
    float*    g2    = h_f32;              // reuse (free after last GAT layer)

    const int B256 = 256;
    const int nh   = N * HDIM;
    const int nEH  = (E + N) * HEADS;
    dim3 gemmGrid((N + 15) / 16, 1, 1);

    // ---- input projection ----
    input_proj_kernel<<<(nh + B256 - 1) / B256, B256, 0, stream>>>(
        x, W_in, b_in, h_f32, h_f16, N);

    // ---- 3 GAT layers ----
    for (int l = 0; l < 3; ++l) {
        cvt16_swz_kernel<<<(128 * 128 + B256 - 1) / B256, B256, 0, stream>>>(
            gatW[l], Wh, 128, 128);
        gemm_wmma_kernel<128><<<gemmGrid, B256, 0, stream>>>(h_f16, Wh, nullptr, h2, N, 0);
        alpha_kernel<<<(N * HEADS + B256 - 1) / B256, B256, 0, stream>>>(
            h2, gatAS[l], gatAD[l], asb, adb, N);
        init_maxv_kernel<<<(N * HEADS + B256 - 1) / B256, B256, 0, stream>>>(maxv, N * HEADS);
        hipMemsetAsync(denom, 0, (size_t)N * HEADS * 4, stream);
        hipMemsetAsync(agg,   0, (size_t)N * HDIM * 4, stream);
        edge_max_kernel<<<(nEH + B256 - 1) / B256, B256, 0, stream>>>(ei, E, N, asb, adb, maxv);
        edge_sum_kernel<<<(nEH + B256 - 1) / B256, B256, 0, stream>>>(ei, E, N, asb, adb, maxv, denom);
        edge_scatter_kernel<<<(nEH + B256 - 1) / B256, B256, 0, stream>>>(
            ei, E, N, asb, adb, maxv, denom, h2, agg);
        post_kernel<<<(N + 7) / 8, B256, 0, stream>>>(
            agg, gatB[l], lnG[l], lnB[l], h_f32, h_f16, N);
    }

    // ---- MLP head ----
    cvt16_swz_kernel<<<(128 * 128 + B256 - 1) / B256, B256, 0, stream>>>(W1, Wh, 128, 128);
    gemm_wmma_kernel<128><<<gemmGrid, B256, 0, stream>>>(h_f16, Wh, b1, h2, N, 1);
    cvt16_kernel<<<(nh + B256 - 1) / B256, B256, 0, stream>>>(h2, g1h, nh);
    cvt16_swz_kernel<<<(128 * 64 + B256 - 1) / B256, B256, 0, stream>>>(W2, Wh, 128, 64);
    gemm_wmma_kernel<64><<<gemmGrid, B256, 0, stream>>>(g1h, Wh, b2, g2, N, 1);
    head_kernel<<<(N + B256 - 1) / B256, B256, 0, stream>>>(g2, W3, b3, (float*)d_out, N);
}
